// MAB_43920335568910
// MI455X (gfx1250) — matrix operand order repaired and checked by reading, not compiled, not run on hardware
//
#include <hip/hip_runtime.h>

// ---------------------------------------------------------------------------
// MAB (Set-Transformer attention block) for gfx1250 / MI455X.
// f16 WMMA 16x16x32 with f32 accumulation, flash-attention streaming softmax.
// Round 2: branch-free hot loops (templated dual-output projection, wrapped
// prefetch index) to kill the v_mov_b64 accumulator shuffling seen in round 1.
// ---------------------------------------------------------------------------

typedef _Float16 half_t;
typedef __attribute__((ext_vector_type(16))) _Float16 v16h;
typedef __attribute__((ext_vector_type(8)))  _Float16 v8h;
typedef __attribute__((ext_vector_type(8)))  float    v8f;

#define DEV __device__ __forceinline__

DEV v8f wmma16x16x32(v16h a, v16h b, v8f c) {
  // D = A(16x32 f16) * B(32x16 f16) + C(16x16 f32)
  return __builtin_amdgcn_wmma_f32_16x16x32_f16(false, a, false, b,
                                                (short)0, c, false, false);
}
DEV v8h ld8(const half_t* p) { return *(const v8h*)p; }
DEV v16h cat16(v8h lo, v8h hi) {
  v16h r;
#pragma unroll
  for (int i = 0; i < 8; ++i) { r[i] = lo[i]; r[i + 8] = hi[i]; }
  return r;
}
// A fragment 16x32: lo-lanes hold K{0..7,16..23}, hi-lanes K{8..15,24..31}.
// rowbase = &lds[row_m * stride + kchunk]; stride (halves) multiple of 8.
DEV v16h frag_a(const half_t* rowbase, int hi) {
  return cat16(ld8(rowbase + hi * 8), ld8(rowbase + hi * 8 + 16));
}
// B fragment 32x16 from column-major LDS: lo-lanes hold K{0..7,16..23},
// hi-lanes K{8..15,24..31}. colbase = &lds[col_n * stride + kchunk].
DEV v16h frag_b(const half_t* colbase, int hi) {
  return cat16(ld8(colbase + hi * 8), ld8(colbase + hi * 8 + 16));
}

// ---------------------------------------------------------------------------
// Projection GEMM: out = A[M,512](f32) @ W[512,512](f32) + b. DUAL variant
// computes a second product sharing the A staging (k and v from the same K).
// 128 threads = 4 waves; block tile 64(M)x64(N); K chunk 32. Branch-free loop.
// ---------------------------------------------------------------------------
template <bool DUAL, bool OUT_F32>
__global__ __launch_bounds__(128)
void proj_kernel(const float* __restrict__ A,
                 const float* __restrict__ W1, const float* __restrict__ b1,
                 half_t* __restrict__ o1h, float* __restrict__ o1f,
                 const float* __restrict__ W2, const float* __restrict__ b2,
                 half_t* __restrict__ o2h) {
  __shared__ half_t ldsA[64 * 40];   // [row][k]  stride 40 (80B, 16B-aligned)
  __shared__ half_t ldsB1[64 * 40];  // [n][k] column-major
  __shared__ half_t ldsB2[DUAL ? 64 * 40 : 8];
  const int t = threadIdx.x;
  const int lane = t & 31, wv = t >> 5;
  const int hi = lane >> 4, n = lane & 15;
  const int row0 = blockIdx.x * 64;
  const int col0 = blockIdx.y * 64;

  v8f acc1[4] = {}; v8f acc2[4] = {};
  for (int kb = 0; kb < 16; ++kb) {
#pragma unroll
    for (int i = 0; i < 16; ++i) {                     // stage A 64x32, cvt f16
      int idx = t + i * 128;
      int r = idx >> 5, c = idx & 31;
      ldsA[r * 40 + c] = (half_t)A[(size_t)(row0 + r) * 512 + kb * 32 + c];
    }
#pragma unroll
    for (int i = 0; i < 16; ++i) {                     // stage W 32x64 -> [n][k]
      int idx = t + i * 128;
      int k = idx >> 6, c = idx & 63;
      size_t g = (size_t)(kb * 32 + k) * 512 + col0 + c;
      ldsB1[c * 40 + k] = (half_t)W1[g];
      if (DUAL) ldsB2[c * 40 + k] = (half_t)W2[g];
    }
    __syncthreads();
    v16h af = frag_a(&ldsA[(wv * 16 + n) * 40], hi);
    v16h bf1[4];
#pragma unroll
    for (int nt = 0; nt < 4; ++nt)
      bf1[nt] = frag_b(&ldsB1[(nt * 16 + n) * 40], hi);
#pragma unroll
    for (int nt = 0; nt < 4; ++nt)
      acc1[nt] = wmma16x16x32(af, bf1[nt], acc1[nt]);
    if (DUAL) {
      v16h bf2[4];
#pragma unroll
      for (int nt = 0; nt < 4; ++nt)
        bf2[nt] = frag_b(&ldsB2[(nt * 16 + n) * 40], hi);
#pragma unroll
      for (int nt = 0; nt < 4; ++nt)
        acc2[nt] = wmma16x16x32(af, bf2[nt], acc2[nt]);
    }
    __syncthreads();
  }
#pragma unroll
  for (int nt = 0; nt < 4; ++nt) {
#pragma unroll
    for (int r = 0; r < 8; ++r) {                      // C layout: M=r+8*hi
      int row = row0 + wv * 16 + r + hi * 8;
      int col = col0 + nt * 16 + n;
      size_t g = (size_t)row * 512 + col;
      float v1 = acc1[nt][r] + b1[col];
      if (OUT_F32) o1f[g] = v1;
      else         o1h[g] = (half_t)v1;
      if (DUAL)    o2h[g] = (half_t)(acc2[nt][r] + b2[col]);
    }
  }
}

// ---------------------------------------------------------------------------
// Flash attention: one workgroup = 64 query rows for one (h,b); streams 64-key
// chunks of K/V through LDS; online softmax; O = q + (P@V)/l  (residual).
// Wave w owns rows [w*16, w*16+16); Q A-fragments live in registers.
// ---------------------------------------------------------------------------
__global__ __launch_bounds__(128)
void attn_kernel(const float* __restrict__ qf, const half_t* __restrict__ kf,
                 const half_t* __restrict__ vf, const float* __restrict__ mask,
                 float* __restrict__ O) {
  __shared__ half_t ldsQ[64 * 72];   // [qrow][d]   stride 72 (144B, 16B-align)
  __shared__ half_t ldsK[64 * 72];   // [key][d]    (column-major for S's B op)
  __shared__ half_t ldsV[64 * 72];   // [d][key]    (column-major for PV's B op)
  __shared__ half_t ldsP[64 * 72];   // [qrow][key] per-wave private rows
  const int t = threadIdx.x;
  const int lane = t & 31, wv = t >> 5;
  const int hi = lane >> 4, n = lane & 15;
  const int q0 = blockIdx.x * 64;
  const int h = blockIdx.y;
  const int b = blockIdx.z;
  const float scale = 0.044194173824159216f;  // 1/sqrt(512) (== 1/sqrt(DV))

#pragma unroll
  for (int i = 0; i < 32; ++i) {               // stage Q tile (f32 -> f16)
    int idx = t + i * 128;
    int r = idx >> 6, d = idx & 63;
    ldsQ[r * 72 + d] = (half_t)qf[(size_t)(b * 1024 + q0 + r) * 512 + h * 64 + d];
  }
  __syncthreads();
  v16h aq[2];
  aq[0] = frag_a(&ldsQ[(wv * 16 + n) * 72 + 0], hi);
  aq[1] = frag_a(&ldsQ[(wv * 16 + n) * 72 + 32], hi);

  float mrow[8], lrow[8];
#pragma unroll
  for (int r = 0; r < 8; ++r) { mrow[r] = -3.0e38f; lrow[r] = 0.f; }
  v8f oacc[4] = {};

  for (int kc = 0; kc < 64; ++kc) {
    const size_t kvbase = (size_t)(b * 4096 + kc * 64) * 512 + h * 64;
#pragma unroll
    for (int i = 0; i < 32; ++i) {             // stage K row-major, V transposed
      int idx = t + i * 128;
      int key = idx >> 6, d = idx & 63;
      ldsK[key * 72 + d] = kf[kvbase + (size_t)key * 512 + d];
      ldsV[d * 72 + key] = vf[kvbase + (size_t)key * 512 + d];
    }
    __syncthreads();
    // branch-free prefetch of next chunk (index wraps, always in-bounds)
    {
      const size_t nb = (size_t)(b * 4096 + ((kc + 1) & 63) * 64) * 512 + h * 64;
      __builtin_prefetch(kf + nb, 0, 1);
      __builtin_prefetch(vf + nb, 0, 1);
    }
    // S = Q @ K^T  (4 key tiles x 2 k-chunks of 32)
    v8f s[4] = {};
#pragma unroll
    for (int nt = 0; nt < 4; ++nt) {
      v16h bf0 = frag_b(&ldsK[(nt * 16 + n) * 72 + 0], hi);
      v16h bf1 = frag_b(&ldsK[(nt * 16 + n) * 72 + 32], hi);
      s[nt] = wmma16x16x32(aq[0], bf0, s[nt]);
      s[nt] = wmma16x16x32(aq[1], bf1, s[nt]);
    }
    // scale + mask (C layout: row = r + 8*hi, col = nt*16 + n)
#pragma unroll
    for (int nt = 0; nt < 4; ++nt)
#pragma unroll
      for (int r = 0; r < 8; ++r) {
        int qrow = q0 + wv * 16 + r + hi * 8;
        int key = kc * 64 + nt * 16 + n;
        s[nt][r] = s[nt][r] * scale + mask[(size_t)(b * 1024 + qrow) * 4096 + key];
      }
    // online softmax: rows live in 16-lane halves -> shfl_xor 1,2,4,8
    float alpha[8];
#pragma unroll
    for (int r = 0; r < 8; ++r) {
      float cm = fmaxf(fmaxf(s[0][r], s[1][r]), fmaxf(s[2][r], s[3][r]));
#pragma unroll
      for (int m = 1; m <= 8; m <<= 1) cm = fmaxf(cm, __shfl_xor(cm, m, 32));
      float mn = fmaxf(mrow[r], cm);
      alpha[r] = __expf(mrow[r] - mn);
      mrow[r] = mn;
    }
    float rs[8] = {};
#pragma unroll
    for (int nt = 0; nt < 4; ++nt)
#pragma unroll
      for (int r = 0; r < 8; ++r) {
        float p = __expf(s[nt][r] - mrow[r]);
        rs[r] += p;
        ldsP[(wv * 16 + r + hi * 8) * 72 + nt * 16 + n] = (half_t)p;
      }
#pragma unroll
    for (int r = 0; r < 8; ++r) {
#pragma unroll
      for (int m = 1; m <= 8; m <<= 1) rs[r] += __shfl_xor(rs[r], m, 32);
      lrow[r] = lrow[r] * alpha[r] + rs[r];
    }
#pragma unroll
    for (int dt = 0; dt < 4; ++dt)
#pragma unroll
      for (int r = 0; r < 8; ++r) oacc[dt][r] *= alpha[r];
    // O += P @ V  (same-wave ldsP round-trip re-layouts C -> A operand)
    v16h pa0 = frag_a(&ldsP[(wv * 16 + n) * 72 + 0], hi);
    v16h pa1 = frag_a(&ldsP[(wv * 16 + n) * 72 + 32], hi);
#pragma unroll
    for (int dt = 0; dt < 4; ++dt) {
      v16h bf0 = frag_b(&ldsV[(dt * 16 + n) * 72 + 0], hi);
      v16h bf1 = frag_b(&ldsV[(dt * 16 + n) * 72 + 32], hi);
      oacc[dt] = wmma16x16x32(pa0, bf0, oacc[dt]);
      oacc[dt] = wmma16x16x32(pa1, bf1, oacc[dt]);
    }
    __syncthreads();
  }
  // epilogue: merged-head layout O[b][q][h*64+d] = q + (P@V)/l
#pragma unroll
  for (int dt = 0; dt < 4; ++dt)
#pragma unroll
    for (int r = 0; r < 8; ++r) {
      int qrow = q0 + wv * 16 + r + hi * 8;
      int col = h * 64 + dt * 16 + n;
      size_t g = (size_t)(b * 1024 + qrow) * 512 + col;
      O[g] = qf[g] + oacc[dt][r] / lrow[r];
    }
}

// ---------------------------------------------------------------------------
// Row LayerNorm over 512 cols; one wave per row (16 elems/lane).
// ---------------------------------------------------------------------------
template <bool EMIT_F16>
__global__ __launch_bounds__(256)
void ln_kernel(const float* __restrict__ X, const float* __restrict__ g,
               const float* __restrict__ bta, float* __restrict__ Yf,
               half_t* __restrict__ Yh) {
  const int lane = threadIdx.x & 31, wv = threadIdx.x >> 5;
  const int row = blockIdx.x * 8 + wv;
  const float* xr = X + (size_t)row * 512;
  float x[16];
#pragma unroll
  for (int i = 0; i < 4; ++i) {
    const float4 v = ((const float4*)xr)[lane * 4 + i];
    x[i * 4 + 0] = v.x; x[i * 4 + 1] = v.y; x[i * 4 + 2] = v.z; x[i * 4 + 3] = v.w;
  }
  float s = 0.f, sq = 0.f;
#pragma unroll
  for (int i = 0; i < 16; ++i) { s += x[i]; sq += x[i] * x[i]; }
#pragma unroll
  for (int m = 1; m <= 16; m <<= 1) {
    s += __shfl_xor(s, m, 32);
    sq += __shfl_xor(sq, m, 32);
  }
  float mu = s * (1.f / 512.f);
  float var = sq * (1.f / 512.f) - mu * mu;
  float rstd = rsqrtf(var + 1e-5f);
#pragma unroll
  for (int i = 0; i < 16; ++i) {
    int c = lane * 16 + i;
    float y = (x[i] - mu) * rstd * g[c] + bta[c];
    Yf[(size_t)row * 512 + c] = y;
    if (EMIT_F16) Yh[(size_t)row * 512 + c] = (half_t)y;
  }
}

// ---------------------------------------------------------------------------
// Z = X + relu(X @ Wo + bo): same tiling as proj, f32 residual epilogue.
// ---------------------------------------------------------------------------
__global__ __launch_bounds__(128)
void gemm2_kernel(const half_t* __restrict__ Ah, const float* __restrict__ Af,
                  const float* __restrict__ W, const float* __restrict__ bias,
                  float* __restrict__ Z) {
  __shared__ half_t ldsA[64 * 40];
  __shared__ half_t ldsB[64 * 40];
  const int t = threadIdx.x;
  const int lane = t & 31, wv = t >> 5;
  const int hi = lane >> 4, n = lane & 15;
  const int row0 = blockIdx.x * 64;
  const int col0 = blockIdx.y * 64;
  v8f acc[4] = {};
  for (int kb = 0; kb < 16; ++kb) {
#pragma unroll
    for (int i = 0; i < 16; ++i) {
      int idx = t + i * 128;
      int r = idx >> 5, c = idx & 31;
      ldsA[r * 40 + c] = Ah[(size_t)(row0 + r) * 512 + kb * 32 + c];
    }
#pragma unroll
    for (int i = 0; i < 16; ++i) {
      int idx = t + i * 128;
      int k = idx >> 6, c = idx & 63;
      ldsB[c * 40 + k] = (half_t)W[(size_t)(kb * 32 + k) * 512 + col0 + c];
    }
    __syncthreads();
    v16h af = frag_a(&ldsA[(wv * 16 + n) * 40], hi);
    v16h bf[4];
#pragma unroll
    for (int nt = 0; nt < 4; ++nt)
      bf[nt] = frag_b(&ldsB[(nt * 16 + n) * 40], hi);
#pragma unroll
    for (int nt = 0; nt < 4; ++nt)
      acc[nt] = wmma16x16x32(af, bf[nt], acc[nt]);
    __syncthreads();
  }
#pragma unroll
  for (int nt = 0; nt < 4; ++nt)
#pragma unroll
    for (int r = 0; r < 8; ++r) {
      int row = row0 + wv * 16 + r + hi * 8;
      int col = col0 + nt * 16 + n;
      size_t g = (size_t)row * 512 + col;
      Z[g] = Af[g] + fmaxf(acc[nt][r] + bias[col], 0.f);
    }
}

// ---------------------------------------------------------------------------
extern "C" void kernel_launch(void* const* d_in, const int* in_sizes, int n_in,
                              void* d_out, int out_size, void* d_ws, size_t ws_size,
                              hipStream_t stream) {
  const float* Q    = (const float*)d_in[0];
  const float* Kx   = (const float*)d_in[1];
  const float* mask = (const float*)d_in[2];
  const float* Wq   = (const float*)d_in[3];
  const float* bq   = (const float*)d_in[4];
  const float* Wk   = (const float*)d_in[5];
  const float* bk   = (const float*)d_in[6];
  const float* Wv   = (const float*)d_in[7];
  const float* bv   = (const float*)d_in[8];
  const float* Wo   = (const float*)d_in[9];
  const float* bo   = (const float*)d_in[10];
  const float* g0   = (const float*)d_in[11];
  const float* b0   = (const float*)d_in[12];
  const float* g1   = (const float*)d_in[13];
  const float* b1   = (const float*)d_in[14];
  float* out = (float*)d_out;

  char* w = (char*)d_ws;  // workspace layout (60 MB total)
  float*  q_f32 = (float*)(w);                     //  8 MB  [4096,512] f32
  half_t* k_f16 = (half_t*)(w + (8ull << 20));     // 16 MB  [16384,512] f16
  half_t* v_f16 = (half_t*)(w + (24ull << 20));    // 16 MB  [16384,512] f16
  float*  O_f32 = (float*)(w + (40ull << 20));     //  8 MB  attn out / Z (reused)
  float*  X_f32 = (float*)(w + (48ull << 20));     //  8 MB  post-LN0
  half_t* X_f16 = (half_t*)(w + (56ull << 20));    //  4 MB  post-LN0 f16
  float*  Z_f32 = O_f32;                           //  overlay: O dead after LN0

  // q = Q@Wq + bq  (kept f32 for the attention residual)
  proj_kernel<false, true><<<dim3(64, 8), 128, 0, stream>>>(
      Q, Wq, bq, nullptr, q_f32, nullptr, nullptr, nullptr);
  // k = K@Wk + bk, v = K@Wv + bv (fused: one pass over K)
  proj_kernel<true, false><<<dim3(256, 8), 128, 0, stream>>>(
      Kx, Wk, bk, k_f16, nullptr, Wv, bv, v_f16);
  // flash attention with residual: O = q + softmax(qk^T/sqrt(512)+mask) v
  attn_kernel<<<dim3(16, 8, 4), 128, 0, stream>>>(q_f32, k_f16, v_f16, mask, O_f32);
  // X = LN0(O)
  ln_kernel<true><<<512, 256, 0, stream>>>(O_f32, g0, b0, X_f32, X_f16);
  // Z = X + relu(X@Wo + bo)
  gemm2_kernel<<<dim3(64, 8), 128, 0, stream>>>(X_f16, X_f32, Wo, bo, Z_f32);
  // out = LN1(Z)
  ln_kernel<false><<<512, 256, 0, stream>>>(Z_f32, g1, b1, out, nullptr);
}
